// MambaBlock_30305289241149
// MI455X (gfx1250) — matrix operand order repaired and checked by reading, not compiled, not run on hardware
//
#include <hip/hip_runtime.h>
#include <math.h>
#include <stdint.h>

// ---- problem constants ----
#define DIMC   192
#define LSEQ   4096
#define BATCH  4
#define DI     384          // d_inner
#define DSTATE 16
#define DTRANK 12
#define NPROJ  (2*DI)       // 768
#define XDBL   (DTRANK + 2*DSTATE)  // 44
#define CHUNK  64
#define NCHUNK (LSEQ/CHUNK) // 64
#define BM     128          // GEMM block rows (8 waves x 16)

typedef _Float16 half16 __attribute__((ext_vector_type(16)));
typedef _Float16 hh8    __attribute__((ext_vector_type(8)));
typedef float    floatx8 __attribute__((ext_vector_type(8)));

union H16 { half16 v; hh8 h[2]; _Float16 e[16]; };
union F8  { floatx8 v; float   e[8];  };

// ---- CDNA5 async global->LDS copy (ASYNCcnt-tracked), ISA §10/§15.18 ----
__device__ inline void async_copy_b128(unsigned lds_off, const void* gptr) {
  asm volatile("global_load_async_to_lds_b128 %0, %1, off"
               :: "v"(lds_off), "v"((unsigned long long)(uintptr_t)gptr)
               : "memory");
}
__device__ inline void wait_async0() {
  asm volatile("s_wait_asynccnt 0x0" ::: "memory");
}

// ---- WMMA operand gathers from LDS tiles (aligned 16B ds_load_b128) ----
// A tile row-major [m][32] f16. ISA 16-bit A 16x32 layout:
// lanes 0-15: M=lane, K={0..7,16..23}; lanes 16-31: M=lane-16, K={8..15,24..31}
__device__ inline half16 gather_a(const _Float16* t, int lane) {
  int g = lane >> 4, m = lane & 15;
  const hh8* row = (const hh8*)(t + m * 32);
  H16 a;
  a.h[0] = row[g];        // K = g*8 .. g*8+7
  a.h[1] = row[2 + g];    // K = 16+g*8 .. 16+g*8+7
  return a.v;
}
// B tile stored [n][32] f16 (k contiguous). ISA 16-bit B 32x16 layout:
// lanes 0-15: N=lane, K={0..7,16..23}; lanes 16-31: N=lane-16, K={8..15,24..31}
__device__ inline half16 gather_b(const _Float16* t, int lane) {
  int g = lane >> 4, n = lane & 15;
  const hh8* row = (const hh8*)(t + n * 32);
  H16 b;
  b.h[0] = row[g];
  b.h[1] = row[2 + g];
  return b.v;
}

__device__ inline float fast_silu(float v) {
  // v * sigmoid(v) with v_rcp_f32 (tolerance does not need IEEE divide)
  return v * __builtin_amdgcn_rcpf(1.f + __expf(-v));
}

// ==================== K1: in_proj GEMM, 128x64 block, double-buffered LDS ====
// xz[b,l,e] = sum_d x[b,d,l]*W[e,d].  e<384 -> xp (b,DI,L); e>=384 -> silu(z) (b,L,DI)
__global__ __launch_bounds__(256) void k_in_proj(const float* __restrict__ x,
                                                 const float* __restrict__ w,
                                                 float* __restrict__ xp,
                                                 float* __restrict__ zs) {
  __shared__ __align__(16) _Float16 Ash[2][BM][32];   // 16 KB
  __shared__ __align__(16) _Float16 Bsh[2][64][32];   //  8 KB
  const int t = threadIdx.x;
  const int wave = t >> 5, lane = t & 31;
  const int g = lane >> 4, li = lane & 15;
  const int gl0 = blockIdx.y * BM;           // global row (b*L + l) base
  const int b = gl0 >> 12, l0 = gl0 & (LSEQ - 1);
  const int n0 = blockIdx.x * 64;

  F8 acc[4];
#pragma unroll
  for (int ns = 0; ns < 4; ++ns)
#pragma unroll
    for (int i = 0; i < 8; ++i) acc[ns].e[i] = 0.f;

  auto stageA = [&](int p, int kk) {
    int mrow = t & (BM - 1), kb = (t >> 7) * 16;
#pragma unroll
    for (int j = 0; j < 16; ++j) {
      int k = kb + j;
      Ash[p][mrow][k] = (_Float16)x[(size_t)(b * DIMC + kk + k) * LSEQ + l0 + mrow];
    }
  };
  auto stageB = [&](int p, int kk) {
    if (t < 128) {
      int n = t >> 1, kh = (t & 1) * 16;
      const float4* wp = (const float4*)(w + (size_t)(n0 + n) * DIMC + kk + kh);
      float4 q0 = wp[0], q1 = wp[1], q2 = wp[2], q3 = wp[3];
      float qv[16] = {q0.x,q0.y,q0.z,q0.w, q1.x,q1.y,q1.z,q1.w,
                      q2.x,q2.y,q2.z,q2.w, q3.x,q3.y,q3.z,q3.w};
      _Float16* br = &Bsh[p][n][kh];
#pragma unroll
      for (int j = 0; j < 16; ++j) br[j] = (_Float16)qv[j];
    }
  };

  stageA(0, 0);
  stageB(0, 0);
  __syncthreads();
#pragma unroll
  for (int it = 0; it < DIMC / 32; ++it) {
    const int p = it & 1;
    if (it + 1 < DIMC / 32) {          // prefetch next tile into other buffer
      stageA(p ^ 1, (it + 1) * 32);
      stageB(p ^ 1, (it + 1) * 32);
    }
    half16 av = gather_a(&Ash[p][wave * 16][0], lane);
#pragma unroll
    for (int ns = 0; ns < 4; ++ns) {
      half16 bv = gather_b(&Bsh[p][ns * 16][0], lane);
      acc[ns].v = __builtin_amdgcn_wmma_f32_16x16x32_f16(false, av, false, bv,
                                                         (short)0, acc[ns].v, false, false);
    }
    __syncthreads();                   // single barrier per K-step (ping-pong)
  }

  // D layout: column N = li, rows M = g*8 + r
  const int lw = l0 + wave * 16 + g * 8;
#pragma unroll
  for (int ns = 0; ns < 4; ++ns) {
    int c = n0 + ns * 16 + li;
    if (c < DI) {
      float* dst = xp + (size_t)(b * DI + c) * LSEQ + lw;
      float4 p0 = {acc[ns].e[0], acc[ns].e[1], acc[ns].e[2], acc[ns].e[3]};
      float4 p1 = {acc[ns].e[4], acc[ns].e[5], acc[ns].e[6], acc[ns].e[7]};
      *(float4*)dst = p0;
      *(float4*)(dst + 4) = p1;
    } else {
      int d = c - DI;
#pragma unroll
      for (int r = 0; r < 8; ++r)
        zs[((size_t)(b * LSEQ + lw + r)) * DI + d] = fast_silu(acc[ns].e[r]);
    }
  }
}

// ==================== K2: depthwise causal conv1d + bias + silu ====================
__global__ void k_conv_silu(const float* __restrict__ xp, const float* __restrict__ cw,
                            const float* __restrict__ cb, float* __restrict__ xc) {
  size_t idx = (size_t)blockIdx.x * blockDim.x + threadIdx.x;
  int l = (int)(idx % LSEQ);
  int e = (int)((idx / LSEQ) % DI);
  int b = (int)(idx / ((size_t)LSEQ * DI));
  const float* row = xp + (size_t)(b * DI + e) * LSEQ;
  float s = cb[e];
#pragma unroll
  for (int j = 0; j < 4; ++j) {
    int li = l - 3 + j;
    if (li >= 0) s += row[li] * cw[e * 4 + j];
  }
  xc[((size_t)(b * LSEQ + l)) * DI + e] = fast_silu(s);
}

// ==================== K3a: x_proj (44 outputs, K=384, VALU dot) ====================
__global__ void k_xproj(const float* __restrict__ xc, const float* __restrict__ w,
                        float* __restrict__ dtr, float* __restrict__ Bm,
                        float* __restrict__ Cm) {
  size_t idx = (size_t)blockIdx.x * blockDim.x + threadIdx.x;
  int j = (int)(idx % XDBL);
  size_t bl = idx / XDBL;
  const float4* xr = (const float4*)(xc + bl * DI);
  const float4* wr = (const float4*)(w + (size_t)j * DI);
  float s = 0.f;
#pragma unroll 4
  for (int k = 0; k < DI / 4; ++k) {
    float4 a = xr[k], c = wr[k];
    s += a.x * c.x + a.y * c.y + a.z * c.z + a.w * c.w;
  }
  if (j < DTRANK)               dtr[bl * DTRANK + j] = s;
  else if (j < DTRANK + DSTATE) Bm[bl * DSTATE + (j - DTRANK)] = s;
  else                          Cm[bl * DSTATE + (j - DTRANK - DSTATE)] = s;
}

// ==================== K3b: dt = softplus(dtr @ Wdt^T + b) ====================
__global__ void k_dtproj(const float* __restrict__ dtr, const float* __restrict__ w,
                         const float* __restrict__ bias, float* __restrict__ dt) {
  size_t idx = (size_t)blockIdx.x * blockDim.x + threadIdx.x;
  int d = (int)(idx % DI);
  size_t bl = idx / DI;
  float s = bias[d];
#pragma unroll
  for (int r = 0; r < DTRANK; ++r) s += dtr[bl * DTRANK + r] * w[d * DTRANK + r];
  dt[bl * DI + d] = (s > 20.f) ? s : log1pf(__expf(s));  // softplus
}

// ==================== K4: chunked diagonal selective scan ====================
// pass1: one block per (b,chunk), 384 threads (one per d). B rows async-staged to LDS.
__global__ __launch_bounds__(384) void k_scan1(const float* __restrict__ dt,
                                               const float* __restrict__ xc,
                                               const float* __restrict__ A_log,
                                               const float* __restrict__ Bm,
                                               float* __restrict__ wsA,
                                               float* __restrict__ wsH) {
  __shared__ __align__(16) float Bsh[CHUNK][DSTATE];  // 4 KB
  const int d = threadIdx.x;
  const int chunk = blockIdx.x & (NCHUNK - 1), b = blockIdx.x >> 6;
  const int l0 = chunk * CHUNK;
  const size_t bl0 = (size_t)b * LSEQ + l0;

  // async-stage the chunk's B rows (64x16 f32 contiguous) into LDS
  {
    const float* src = Bm + bl0 * DSTATE;
    unsigned base = (unsigned)(uintptr_t)&Bsh[0][0];
    for (int i = d; i < (CHUNK * DSTATE) / 4; i += 384)
      async_copy_b128(base + i * 16, src + i * 4);
    wait_async0();
  }
  __syncthreads();

  float An[DSTATE];
#pragma unroll
  for (int n = 0; n < DSTATE; ++n) An[n] = -__expf(A_log[d * DSTATE + n]);
  float ap[DSTATE], hp[DSTATE];
#pragma unroll
  for (int n = 0; n < DSTATE; ++n) { ap[n] = 1.f; hp[n] = 0.f; }

  for (int t = 0; t < CHUNK; ++t) {
    size_t bl = bl0 + t;
    float dtv = dt[bl * DI + d];
    float xv  = xc[bl * DI + d];
    const float* Bv = &Bsh[t][0];
#pragma unroll
    for (int n = 0; n < DSTATE; ++n) {
      float aa = __expf(dtv * An[n]);
      hp[n] = aa * hp[n] + dtv * Bv[n] * xv;
      ap[n] *= aa;
    }
  }
  float* oA = wsA + ((((size_t)(b * NCHUNK + chunk)) * DI + d) << 4);
  float* oH = wsH + ((((size_t)(b * NCHUNK + chunk)) * DI + d) << 4);
#pragma unroll
  for (int n = 0; n < DSTATE; ++n) { oA[n] = ap[n]; oH[n] = hp[n]; }
}

// pass2: per (b,d,n) scan the 64 chunk summaries; wsH[c] becomes h-init of chunk c
__global__ void k_scan2(const float* __restrict__ wsA, float* __restrict__ wsH) {
  size_t idx = (size_t)blockIdx.x * blockDim.x + threadIdx.x;
  int n = (int)(idx % DSTATE);
  int d = (int)((idx / DSTATE) % DI);
  int b = (int)(idx / ((size_t)DSTATE * DI));
  float run = 0.f;
  for (int c = 0; c < NCHUNK; ++c) {
    size_t o = ((((size_t)(b * NCHUNK + c)) * DI + d) << 4) + n;
    float ap = wsA[o], h = wsH[o];
    wsH[o] = run;            // state entering chunk c
    run = ap * run + h;
  }
}

// pass3: recompute within chunk from h-init, contract with C, + D*xc, gate silu(z)
__global__ __launch_bounds__(384) void k_scan3(const float* __restrict__ dt,
                                               const float* __restrict__ xc,
                                               const float* __restrict__ A_log,
                                               const float* __restrict__ Bm,
                                               const float* __restrict__ Cm,
                                               const float* __restrict__ Dp,
                                               const float* __restrict__ zs,
                                               const float* __restrict__ wsH,
                                               float* __restrict__ y) {
  __shared__ __align__(16) float Bsh[CHUNK][DSTATE];  // 4 KB
  __shared__ __align__(16) float Csh[CHUNK][DSTATE];  // 4 KB
  const int d = threadIdx.x;
  const int chunk = blockIdx.x & (NCHUNK - 1), b = blockIdx.x >> 6;
  const int l0 = chunk * CHUNK;
  const size_t bl0 = (size_t)b * LSEQ + l0;

  {
    const int NQ = (CHUNK * DSTATE) / 4;  // 256 float4 per matrix
    const float* srcB = Bm + bl0 * DSTATE;
    const float* srcC = Cm + bl0 * DSTATE;
    unsigned baseB = (unsigned)(uintptr_t)&Bsh[0][0];
    unsigned baseC = (unsigned)(uintptr_t)&Csh[0][0];
    for (int i = d; i < 2 * NQ; i += 384) {
      if (i < NQ) async_copy_b128(baseB + i * 16, srcB + i * 4);
      else        async_copy_b128(baseC + (i - NQ) * 16, srcC + (size_t)(i - NQ) * 4);
    }
    wait_async0();
  }
  __syncthreads();

  float An[DSTATE];
#pragma unroll
  for (int n = 0; n < DSTATE; ++n) An[n] = -__expf(A_log[d * DSTATE + n]);
  float h[DSTATE];
  const float* iH = wsH + ((((size_t)(b * NCHUNK + chunk)) * DI + d) << 4);
#pragma unroll
  for (int n = 0; n < DSTATE; ++n) h[n] = iH[n];
  const float Dv = Dp[d];

  for (int t = 0; t < CHUNK; ++t) {
    size_t bl = bl0 + t;
    float dtv = dt[bl * DI + d];
    float xv  = xc[bl * DI + d];
    const float* Bv = &Bsh[t][0];
    const float* Cv = &Csh[t][0];
    float acc = 0.f;
#pragma unroll
    for (int n = 0; n < DSTATE; ++n) {
      float aa = __expf(dtv * An[n]);
      h[n] = aa * h[n] + dtv * Bv[n] * xv;
      acc += h[n] * Cv[n];
    }
    acc += Dv * xv;
    y[bl * DI + d] = acc * zs[bl * DI + d];   // zs already holds silu(z)
  }
}

// ==================== K5: out_proj GEMM, 128x64 block, double-buffered LDS ====
__global__ __launch_bounds__(256) void k_out_proj(const float* __restrict__ y,
                                                  const float* __restrict__ w,
                                                  float* __restrict__ out) {
  __shared__ __align__(16) _Float16 Ash[2][BM][32];   // 16 KB
  __shared__ __align__(16) _Float16 Bsh[2][64][32];   //  8 KB
  const int t = threadIdx.x;
  const int wave = t >> 5, lane = t & 31;
  const int g = lane >> 4, li = lane & 15;
  const int gl0 = blockIdx.y * BM;
  const int b = gl0 >> 12, l0 = gl0 & (LSEQ - 1);
  const int n0 = blockIdx.x * 64;

  F8 acc[4];
#pragma unroll
  for (int ns = 0; ns < 4; ++ns)
#pragma unroll
    for (int i = 0; i < 8; ++i) acc[ns].e[i] = 0.f;

  auto stageA = [&](int p, int kk) {
    int mrow = t >> 1, kh = (t & 1) * 16;   // 256 threads cover 128 rows x 2 halves
    const float4* yp = (const float4*)(y + (size_t)(gl0 + mrow) * DI + kk + kh);
    float4 q0 = yp[0], q1 = yp[1], q2 = yp[2], q3 = yp[3];
    float qv[16] = {q0.x,q0.y,q0.z,q0.w, q1.x,q1.y,q1.z,q1.w,
                    q2.x,q2.y,q2.z,q2.w, q3.x,q3.y,q3.z,q3.w};
    _Float16* ar = &Ash[p][mrow][kh];
#pragma unroll
    for (int j = 0; j < 16; ++j) ar[j] = (_Float16)qv[j];
  };
  auto stageB = [&](int p, int kk) {
    if (t < 128) {
      int n = t >> 1, kh = (t & 1) * 16;
      const float4* wp = (const float4*)(w + (size_t)(n0 + n) * DI + kk + kh);
      float4 q0 = wp[0], q1 = wp[1], q2 = wp[2], q3 = wp[3];
      float qv[16] = {q0.x,q0.y,q0.z,q0.w, q1.x,q1.y,q1.z,q1.w,
                      q2.x,q2.y,q2.z,q2.w, q3.x,q3.y,q3.z,q3.w};
      _Float16* br = &Bsh[p][n][kh];
#pragma unroll
      for (int j = 0; j < 16; ++j) br[j] = (_Float16)qv[j];
    }
  };

  stageA(0, 0);
  stageB(0, 0);
  __syncthreads();
  for (int it = 0; it < DI / 32; ++it) {
    const int p = it & 1;
    if (it + 1 < DI / 32) {
      stageA(p ^ 1, (it + 1) * 32);
      stageB(p ^ 1, (it + 1) * 32);
    }
    half16 av = gather_a(&Ash[p][wave * 16][0], lane);
#pragma unroll
    for (int ns = 0; ns < 4; ++ns) {
      half16 bv = gather_b(&Bsh[p][ns * 16][0], lane);
      acc[ns].v = __builtin_amdgcn_wmma_f32_16x16x32_f16(false, av, false, bv,
                                                         (short)0, acc[ns].v, false, false);
    }
    __syncthreads();
  }
  // out (b, DIMC, L): lane's column c = n0+ns*16+li, 8 consecutive l contiguous
  const int lw = l0 + wave * 16 + g * 8;
#pragma unroll
  for (int ns = 0; ns < 4; ++ns) {
    int c = n0 + ns * 16 + li;
    float* dst = out + (size_t)(b * DIMC + c) * LSEQ + lw;
    float4 p0 = {acc[ns].e[0], acc[ns].e[1], acc[ns].e[2], acc[ns].e[3]};
    float4 p1 = {acc[ns].e[4], acc[ns].e[5], acc[ns].e[6], acc[ns].e[7]};
    *(float4*)dst = p0;
    *(float4*)(dst + 4) = p1;
  }
}

// ==================== host launch ====================
extern "C" void kernel_launch(void* const* d_in, const int* in_sizes, int n_in,
                              void* d_out, int out_size, void* d_ws, size_t ws_size,
                              hipStream_t stream) {
  const float* x         = (const float*)d_in[0];
  const float* in_proj_w = (const float*)d_in[1];
  const float* conv_w    = (const float*)d_in[2];
  const float* conv_b    = (const float*)d_in[3];
  const float* x_proj_w  = (const float*)d_in[4];
  const float* dt_proj_w = (const float*)d_in[5];
  const float* dt_proj_b = (const float*)d_in[6];
  const float* A_log     = (const float*)d_in[7];
  const float* D_param   = (const float*)d_in[8];
  const float* out_proj_w= (const float*)d_in[9];
  float* out = (float*)d_out;

  const size_t nBLD = (size_t)BATCH * LSEQ * DI;                 // 6,291,456
  char* ws = (char*)d_ws;
  size_t off = 0;
  auto take = [&](size_t nfloats) -> float* {
    float* p = (float*)(ws + off);
    off = (off + nfloats * sizeof(float) + 255) & ~(size_t)255;
    return p;
  };
  float* xp  = take(nBLD);                       // (b, DI, L) — reused as y later
  float* zs  = take(nBLD);                       // silu(z), (b, L, DI)
  float* xc  = take(nBLD);                       // (b, L, DI)
  float* dt  = take(nBLD);                       // (b, L, DI)
  float* dtr = take((size_t)BATCH * LSEQ * DTRANK);
  float* Bm  = take((size_t)BATCH * LSEQ * DSTATE);
  float* Cm  = take((size_t)BATCH * LSEQ * DSTATE);
  float* wsA = take((size_t)BATCH * NCHUNK * DI * DSTATE);
  float* wsH = take((size_t)BATCH * NCHUNK * DI * DSTATE);
  float* ybuf = xp;  // xp dead after conv; reuse for y

  // K1: in_proj WMMA GEMM, 128x64 tiles: grid (N/64, M/128)
  k_in_proj<<<dim3(NPROJ / 64, (BATCH * LSEQ) / BM), dim3(256), 0, stream>>>(x, in_proj_w, xp, zs);
  // K2: depthwise causal conv + silu
  k_conv_silu<<<dim3((unsigned)(nBLD / 256)), dim3(256), 0, stream>>>(xp, conv_w, conv_b, xc);
  // K3: x_proj and dt_proj
  k_xproj<<<dim3((unsigned)(((size_t)BATCH * LSEQ * XDBL) / 256)), dim3(256), 0, stream>>>(
      xc, x_proj_w, dtr, Bm, Cm);
  k_dtproj<<<dim3((unsigned)(nBLD / 256)), dim3(256), 0, stream>>>(dtr, dt_proj_w, dt_proj_b, dt);
  // K4: chunked scan — one block per (b, chunk) for passes 1/3
  k_scan1<<<dim3(BATCH * NCHUNK), dim3(384), 0, stream>>>(dt, xc, A_log, Bm, wsA, wsH);
  k_scan2<<<dim3((unsigned)((size_t)BATCH * DI * DSTATE / 256)), dim3(256), 0, stream>>>(wsA, wsH);
  k_scan3<<<dim3(BATCH * NCHUNK), dim3(384), 0, stream>>>(dt, xc, A_log, Bm, Cm, D_param,
                                                          zs, wsH, ybuf);
  // K5: out_proj WMMA GEMM, 128x64 tiles: grid (N/64, M/128)
  k_out_proj<<<dim3(DIMC / 64, (BATCH * LSEQ) / BM), dim3(256), 0, stream>>>(ybuf, out_proj_w, out);
}
